// LRUEmbedding_77541339562280
// MI455X (gfx1250) — compile-verified, matrix operand order checked by reading, not run on hardware
//
#include <hip/hip_runtime.h>
#include <math.h>

// ---------------- problem constants ----------------
#define NUM_LAYERS 4
#define HID 512
#define ST  256
#define DIN 128
#define DOUT 64
#define BATCH 16
#define SEQ 2048
#define BL (BATCH * SEQ)        // 32768 rows
#define CHUNK 256
#define NCHUNK (SEQ / CHUNK)    // 8

// ---------------- GEMM tiling ----------------
#define BM 128
#define BN 128
#define BK 64                   // two WMMA k-steps per LDS stage
#define LDT 72                  // padded LDS stride (elements): 144B rows, 16B-aligned chunks

typedef __attribute__((ext_vector_type(16))) __bf16 v16bf;
typedef __attribute__((ext_vector_type(8)))  float  v8f;

// Async global->LDS path (gfx1250 GLOBAL_LOAD_ASYNC_TO_LDS_B128, ASYNCcnt),
// hedged on builtin availability so the file compiles on any toolchain.
#if defined(__HIP_DEVICE_COMPILE__) && \
    __has_builtin(__builtin_amdgcn_global_load_async_to_lds_b128) && \
    __has_builtin(__builtin_amdgcn_s_wait_asynccnt)
#define HAVE_ASYNC 1
// exact parameter types per hipcc diagnostic: vector_size(16) int, AS1 / AS3
typedef int b128_t __attribute__((vector_size(16)));
typedef __attribute__((address_space(1))) b128_t* gas_b128_p;
typedef __attribute__((address_space(3))) b128_t* las_b128_p;
#else
#define HAVE_ASYNC 0
#endif

// epilogue modes
//  0: out_bf16 = acc + bias[n]                                     (embedding)
//  1: out_f32  = acc                                               (Bu projection)
//  2: out_bf16 = gelu(acc + aux[n] * hin[row,n])                   (C-proj + D-term + GELU)
//  3: out_f32  = acc + bias[n]                                     (MLP branch 1 -> u)
//  4: out_bf16 = aux[row,n] * sigmoid(acc + bias[n]) + hin[row,n]  (GLU + residual)
//  5: out_f32  = acc + bias[n]                                     (final projection, GUARDed N=64)
template <int MODE, bool GUARD>
__global__ __launch_bounds__(256) void gemm_wmma(
    const __bf16* __restrict__ A,      // M x K row-major (bf16)
    const __bf16* __restrict__ Wt,     // N x K row-major (bf16)
    void* __restrict__ out,
    const float* __restrict__ bias,
    const float* __restrict__ aux,
    const __bf16* __restrict__ hin,
    int N, int K)
{
    __shared__ __align__(16) __bf16 sA[2][BM * LDT];
    __shared__ __align__(16) __bf16 sB[2][BN * LDT];

    const int tid  = threadIdx.x;
    const int lane = tid & 31;
    const int wave = tid >> 5;          // 0..7
    const int wm   = wave >> 2;         // 0..1 -> 64-row strip
    const int wn   = wave & 3;          // 0..3 -> 32-col strip
    const int m0   = blockIdx.y * BM;
    const int n0   = blockIdx.x * BN;
    const int lr   = lane & 15;
    const int lg   = lane >> 4;

    v8f acc[4][2] = {};
    const int nk = K / BK;

    // ---- one K-stage of 16 WMMAs from LDS buffer `cur` ----
    auto compute_stage = [&](int cur) {
        #pragma unroll
        for (int ks = 0; ks < 2; ++ks) {
            v16bf bfrag[2];
            #pragma unroll
            for (int ni = 0; ni < 2; ++ni) {
                int rb = wn * 32 + ni * 16 + lr;
                union { uint4 u[2]; v16bf v; } t;
                const __bf16* p = &sB[cur][rb * LDT + ks * 32 + lg * 8];
                t.u[0] = *reinterpret_cast<const uint4*>(p);
                t.u[1] = *reinterpret_cast<const uint4*>(p + 16);
                bfrag[ni] = t.v;
            }
            #pragma unroll
            for (int mi = 0; mi < 4; ++mi) {
                int ra = wm * 64 + mi * 16 + lr;
                union { uint4 u[2]; v16bf v; } t;
                const __bf16* p = &sA[cur][ra * LDT + ks * 32 + lg * 8];
                t.u[0] = *reinterpret_cast<const uint4*>(p);
                t.u[1] = *reinterpret_cast<const uint4*>(p + 16);
                #pragma unroll
                for (int ni = 0; ni < 2; ++ni)
                    acc[mi][ni] = __builtin_amdgcn_wmma_f32_16x16x32_bf16(
                        false, t.v, false, bfrag[ni],
                        (short)0, acc[mi][ni], false, false);
            }
        }
    };

#if HAVE_ASYNC
    if constexpr (!GUARD) {
        // ---- async global->LDS double-buffered pipeline ----
        auto aissue = [&](int buf, int k0) {
            #pragma unroll
            for (int p = 0; p < 4; ++p) {
                int c = tid + p * 256;              // 1024 chunks of 8 elems
                int row = c >> 3, kc = (c & 7) * 8;
                __builtin_amdgcn_global_load_async_to_lds_b128(
                    (gas_b128_p)(A + (size_t)(m0 + row) * K + k0 + kc),
                    (las_b128_p)(&sA[buf][row * LDT + kc]), 0, 0);
                __builtin_amdgcn_global_load_async_to_lds_b128(
                    (gas_b128_p)(Wt + (size_t)(n0 + row) * K + k0 + kc),
                    (las_b128_p)(&sB[buf][row * LDT + kc]), 0, 0);
            }
        };
        aissue(0, 0);
        __builtin_amdgcn_s_wait_asynccnt(0);
        __syncthreads();
        for (int kt = 0; kt < nk; ++kt) {
            int cur = kt & 1;
            bool more = (kt + 1 < nk);
            if (more) aissue(cur ^ 1, (kt + 1) * BK);
            compute_stage(cur);
            if (more) __builtin_amdgcn_s_wait_asynccnt(0);
            __syncthreads();
        }
    } else
#endif
    {
        // ---- register-staged double-buffered pipeline ----
        uint4 rga[4], rgb[4];
        auto gload = [&](int k0) {
            #pragma unroll
            for (int p = 0; p < 4; ++p) {
                int c = tid + p * 256;
                int row = c >> 3, kc = (c & 7) * 8;
                rga[p] = *reinterpret_cast<const uint4*>(A + (size_t)(m0 + row) * K + k0 + kc);
                if constexpr (GUARD) {
                    uint4 z; z.x = z.y = z.z = z.w = 0u;
                    rgb[p] = (n0 + row < N)
                        ? *reinterpret_cast<const uint4*>(Wt + (size_t)(n0 + row) * K + k0 + kc)
                        : z;
                } else {
                    rgb[p] = *reinterpret_cast<const uint4*>(Wt + (size_t)(n0 + row) * K + k0 + kc);
                }
            }
        };
        auto sstore = [&](int buf) {
            #pragma unroll
            for (int p = 0; p < 4; ++p) {
                int c = tid + p * 256;
                int row = c >> 3, kc = (c & 7) * 8;
                *reinterpret_cast<uint4*>(&sA[buf][row * LDT + kc]) = rga[p];
                *reinterpret_cast<uint4*>(&sB[buf][row * LDT + kc]) = rgb[p];
            }
        };
        gload(0); sstore(0); __syncthreads();
        for (int kt = 0; kt < nk; ++kt) {
            int cur = kt & 1;
            bool more = (kt + 1 < nk);
            if (more) gload((kt + 1) * BK);
            compute_stage(cur);
            if (more) sstore(cur ^ 1);
            __syncthreads();
        }
    }

    // ---- epilogue ----
    #pragma unroll
    for (int mi = 0; mi < 4; ++mi) {
        #pragma unroll
        for (int ni = 0; ni < 2; ++ni) {
            int gc = n0 + wn * 32 + ni * 16 + lr;
            if (GUARD && gc >= N) continue;
            #pragma unroll
            for (int r = 0; r < 8; ++r) {
                int gr = m0 + wm * 64 + mi * 16 + lg * 8 + r;
                size_t idx = (size_t)gr * N + gc;
                float v = acc[mi][ni][r];
                if constexpr (MODE == 0) {
                    ((__bf16*)out)[idx] = (__bf16)(v + bias[gc]);
                } else if constexpr (MODE == 1) {
                    ((float*)out)[idx] = v;
                } else if constexpr (MODE == 2) {
                    v += aux[gc] * (float)hin[idx];
                    v = 0.5f * v * (1.0f + erff(v * 0.70710678118654752f));
                    ((__bf16*)out)[idx] = (__bf16)v;
                } else if constexpr (MODE == 3) {
                    ((float*)out)[idx] = v + bias[gc];
                } else if constexpr (MODE == 4) {
                    float s = 1.0f / (1.0f + expf(-(v + bias[gc])));
                    ((__bf16*)out)[idx] = (__bf16)(aux[idx] * s + (float)hin[idx]);
                } else {
                    ((float*)out)[idx] = v + bias[gc];
                }
            }
        }
    }
}

// ---------------- chunked complex diagonal scan ----------------
// Bu layout: (BL rows, 512 cols) fp32; col s = real, col 256+s = imag.

__global__ __launch_bounds__(256) void lru_scan_pass1(
    float* __restrict__ Bu, float* __restrict__ carry,
    const float* __restrict__ nu, const float* __restrict__ th)
{
    int s = threadIdx.x;
    int b = blockIdx.x >> 3;
    int c = blockIdx.x & (NCHUNK - 1);
    float r  = expf(-expf(nu[s]));
    float ph = expf(th[s]);
    float Lre = r * cosf(ph), Lim = r * sinf(ph);
    float hre = 0.f, him = 0.f;
    size_t base = ((size_t)b * SEQ + (size_t)c * CHUNK) * (size_t)HID + s;
    for (int t = 0; t < CHUNK; ++t) {
        float re = Bu[base], im = Bu[base + ST];
        float nr = Lre * hre - Lim * him + re;
        float ni = Lre * him + Lim * hre + im;
        hre = nr; him = ni;
        Bu[base] = hre; Bu[base + ST] = him;
        base += HID;
    }
    int ci = b * NCHUNK + c;
    carry[(size_t)ci * HID + s]      = hre;
    carry[(size_t)ci * HID + ST + s] = him;
}

__global__ __launch_bounds__(256) void lru_scan_pass2(
    const float* __restrict__ carry, float* __restrict__ Epre,
    const float* __restrict__ nu, const float* __restrict__ th)
{
    int s = threadIdx.x;
    int b = blockIdx.x;
    float en  = expf(nu[s]);
    float rT  = expf(-((float)CHUNK) * en);
    float phT = ((float)CHUNK) * expf(th[s]);
    float LTre = rT * cosf(phT), LTim = rT * sinf(phT);
    float Ere = 0.f, Eim = 0.f;
    for (int c = 0; c < NCHUNK; ++c) {
        size_t ci = (size_t)(b * NCHUNK + c) * HID;
        Epre[ci + s]      = Ere;
        Epre[ci + ST + s] = Eim;
        float cre = carry[ci + s], cim = carry[ci + ST + s];
        float nr = LTre * Ere - LTim * Eim + cre;
        float ni = LTre * Eim + LTim * Ere + cim;
        Ere = nr; Eim = ni;
    }
}

__global__ __launch_bounds__(256) void lru_scan_pass3(
    const float* __restrict__ Bu, const float* __restrict__ Epre,
    __bf16* __restrict__ hs,
    const float* __restrict__ nu, const float* __restrict__ th)
{
    int s = threadIdx.x;
    int b = blockIdx.x >> 3;
    int c = blockIdx.x & (NCHUNK - 1);
    float r  = expf(-expf(nu[s]));
    float ph = expf(th[s]);
    float Lre = r * cosf(ph), Lim = r * sinf(ph);
    size_t ci = (size_t)(b * NCHUNK + c) * HID;
    float pre = Epre[ci + s], pim = Epre[ci + ST + s];
    size_t base = ((size_t)b * SEQ + (size_t)c * CHUNK) * (size_t)HID + s;
    for (int t = 0; t < CHUNK; ++t) {
        float nr = Lre * pre - Lim * pim;
        float ni = Lre * pim + Lim * pre;
        pre = nr; pim = ni;
        hs[base]      = (__bf16)(Bu[base]      + pre);
        hs[base + ST] = (__bf16)(Bu[base + ST] + pim);
        base += HID;
    }
}

// ---------------- small conversion / packing kernels ----------------
__global__ void conv_f32_bf16(const float* __restrict__ in, __bf16* __restrict__ o, int n) {
    int i = blockIdx.x * blockDim.x + threadIdx.x;
    if (i < n) o[i] = (__bf16)in[i];
}

__global__ void pack_Wb(const float* __restrict__ Bre, const float* __restrict__ Bim,
                        const float* __restrict__ gamma, __bf16* __restrict__ Wb) {
    int i = blockIdx.x * blockDim.x + threadIdx.x;   // over 512*512
    int n = i >> 9, k = i & 511;
    float v;
    if (n < ST) v = Bre[(size_t)n * HID + k] * expf(gamma[n]);
    else        v = Bim[(size_t)(n - ST) * HID + k] * expf(gamma[n - ST]);
    Wb[i] = (__bf16)v;
}

__global__ void pack_Wc(const float* __restrict__ Cre, const float* __restrict__ Cim,
                        __bf16* __restrict__ Wc) {
    int i = blockIdx.x * blockDim.x + threadIdx.x;   // over 512*512
    int n = i >> 9, k = i & 511;
    float v = (k < ST) ? Cre[(size_t)n * ST + k] : -Cim[(size_t)n * ST + (k - ST)];
    Wc[i] = (__bf16)v;
}

// ---------------- host-side orchestration ----------------
extern "C" void kernel_launch(void* const* d_in, const int* in_sizes, int n_in,
                              void* d_out, int out_size, void* d_ws, size_t ws_size,
                              hipStream_t stream) {
    const float* x      = (const float*)d_in[0];
    const float* emb_W  = (const float*)d_in[1];
    const float* emb_b  = (const float*)d_in[2];
    const float* nu_log = (const float*)d_in[3];
    const float* th_log = (const float*)d_in[4];
    const float* ga_log = (const float*)d_in[5];
    const float* B_re   = (const float*)d_in[6];
    const float* B_im   = (const float*)d_in[7];
    const float* C_re   = (const float*)d_in[8];
    const float* C_im   = (const float*)d_in[9];
    const float* Dv     = (const float*)d_in[10];
    const float* W1     = (const float*)d_in[11];
    const float* b1     = (const float*)d_in[12];
    const float* W2     = (const float*)d_in[13];
    const float* b2     = (const float*)d_in[14];
    const float* out_W  = (const float*)d_in[15];
    const float* out_b  = (const float*)d_in[16];
    float* out = (float*)d_out;

    char* ws = (char*)d_ws;
    size_t off = 0;
    auto alloc = [&](size_t bytes) -> void* {
        off = (off + 255) & ~((size_t)255);
        void* p = ws + off;
        off += bytes;
        return p;
    };
    __bf16* x_bf   = (__bf16*)alloc((size_t)BL * DIN * 2);
    __bf16* h_bf   = (__bf16*)alloc((size_t)BL * HID * 2);
    __bf16* hs_bf  = (__bf16*)alloc((size_t)BL * HID * 2);
    __bf16* y_bf   = (__bf16*)alloc((size_t)BL * HID * 2);
    float*  Bu     = (float*) alloc((size_t)BL * HID * 4);   // reused as u
    float*  carry  = (float*) alloc((size_t)BATCH * NCHUNK * HID * 4);
    float*  Epre   = (float*) alloc((size_t)BATCH * NCHUNK * HID * 4);
    __bf16* Wb     = (__bf16*)alloc((size_t)HID * HID * 2);
    __bf16* Wc     = (__bf16*)alloc((size_t)HID * HID * 2);
    __bf16* W1b    = (__bf16*)alloc((size_t)HID * HID * 2);
    __bf16* W2b    = (__bf16*)alloc((size_t)HID * HID * 2);
    __bf16* embWb  = (__bf16*)alloc((size_t)HID * DIN * 2);
    __bf16* outWb  = (__bf16*)alloc((size_t)DOUT * HID * 2);

    auto conv = [&](const float* src, __bf16* dst, int n) {
        conv_f32_bf16<<<(n + 255) / 256, 256, 0, stream>>>(src, dst, n);
    };
    auto gemm = [&](const __bf16* A, const __bf16* Wt, void* o, const float* bias,
                    const float* aux, const __bf16* hin, int N, int K, int mode) {
        dim3 g((N + BN - 1) / BN, BL / BM);
        switch (mode) {
        case 0: gemm_wmma<0, false><<<g, 256, 0, stream>>>(A, Wt, o, bias, aux, hin, N, K); break;
        case 1: gemm_wmma<1, false><<<g, 256, 0, stream>>>(A, Wt, o, bias, aux, hin, N, K); break;
        case 2: gemm_wmma<2, false><<<g, 256, 0, stream>>>(A, Wt, o, bias, aux, hin, N, K); break;
        case 3: gemm_wmma<3, false><<<g, 256, 0, stream>>>(A, Wt, o, bias, aux, hin, N, K); break;
        case 4: gemm_wmma<4, false><<<g, 256, 0, stream>>>(A, Wt, o, bias, aux, hin, N, K); break;
        default: gemm_wmma<5, true><<<g, 256, 0, stream>>>(A, Wt, o, bias, aux, hin, N, K); break;
        }
    };

    // one-time conversions
    conv(x, x_bf, BL * DIN);
    conv(emb_W, embWb, HID * DIN);
    conv(out_W, outWb, DOUT * HID);

    // embedding: h = bf16(x @ emb_W^T + emb_b)
    gemm(x_bf, embWb, h_bf, emb_b, nullptr, nullptr, HID, DIN, 0);

    for (int l = 0; l < NUM_LAYERS; ++l) {
        const float* nu = nu_log + l * ST;
        const float* th = th_log + l * ST;
        pack_Wb<<<(HID * HID) / 256, 256, 0, stream>>>(
            B_re + (size_t)l * ST * HID, B_im + (size_t)l * ST * HID,
            ga_log + l * ST, Wb);
        pack_Wc<<<(HID * HID) / 256, 256, 0, stream>>>(
            C_re + (size_t)l * HID * ST, C_im + (size_t)l * HID * ST, Wc);
        conv(W1 + (size_t)l * HID * HID, W1b, HID * HID);
        conv(W2 + (size_t)l * HID * HID, W2b, HID * HID);

        // Bu = h @ [Bnorm_re ; Bnorm_im]^T   (fp32, packed re|im)
        gemm(h_bf, Wb, Bu, nullptr, nullptr, nullptr, HID, HID, 1);

        // chunked diagonal complex scan over the sequence
        lru_scan_pass1<<<BATCH * NCHUNK, ST, 0, stream>>>(Bu, carry, nu, th);
        lru_scan_pass2<<<BATCH, ST, 0, stream>>>(carry, Epre, nu, th);
        lru_scan_pass3<<<BATCH * NCHUNK, ST, 0, stream>>>(Bu, Epre, hs_bf, nu, th);

        // y = gelu(Re(h_state @ C^T) + h * D)
        gemm(hs_bf, Wc, y_bf, nullptr, Dv + (size_t)l * HID, h_bf, HID, HID, 2);

        // u = y @ W1^T + b1   (fp32, reuse Bu buffer)
        gemm(y_bf, W1b, Bu, b1 + (size_t)l * HID, nullptr, nullptr, HID, HID, 3);
        // h = u * sigmoid(y @ W2^T + b2) + h   (bf16, in place)
        gemm(y_bf, W2b, h_bf, b2 + (size_t)l * HID, Bu, h_bf, HID, HID, 4);
    }

    // final projection: out = h @ out_W^T + out_b (fp32)
    gemm(h_bf, outWb, out, out_b, nullptr, nullptr, DOUT, HID, 5);
}
